// DynamicRouter_61821759259046
// MI455X (gfx1250) — compile-verified, matrix operand order checked by reading, not run on hardware
//
#include <hip/hip_runtime.h>
#include <math.h>

// ---------------- CDNA5 WMMA types ----------------
typedef __attribute__((ext_vector_type(16))) __bf16 v16bf;
typedef __attribute__((ext_vector_type(8)))  float  v8f;

// ---------------- Problem constants ----------------
#define N_TOK 8192   // B*S = 4*2048
#define HDIM  1024
#define NEXP  8
#define FDIM  2048   // 2*H
#define CAP   8192   // top-2 distinct -> each expert gets <= N_TOK tokens
#define TILE_M 32    // tokens per workgroup (2 WMMA M-tiles)

// ---------------- Workspace layout (bytes) ----------------
static constexpr size_t WS_COUNTS = 0;
static constexpr size_t WS_USAGE  = 64;
static constexpr size_t WS_TOK    = 1024;
static constexpr size_t WS_PROB   = WS_TOK + (size_t)NEXP * CAP * 4;
static constexpr size_t WS_W1S    = ((WS_PROB + (size_t)NEXP * CAP * 4) + 4095) & ~(size_t)4095;
static constexpr size_t W1S_FRAGS = (size_t)NEXP * 32 * 128;           // e * kb(H/32) * n(F/16)
static constexpr size_t WS_W2S    = WS_W1S + W1S_FRAGS * 1024;         // 1KB per fragment
static constexpr size_t W2S_FRAGS = (size_t)NEXP * 64 * 64;            // e * kb(F/32) * n(H/16)
// total ws need ~= 64.5 MB

__device__ __forceinline__ __bf16 f2bf(float f) {
    unsigned u = __float_as_uint(f);
    u += 0x7FFFu + ((u >> 16) & 1u);                 // RNE f32 -> bf16
    unsigned short s = (unsigned short)(u >> 16);
    union { unsigned short s; __bf16 b; } c; c.s = s;
    return c.b;
}

#define WMMA_BF16(A, B, C) \
    __builtin_amdgcn_wmma_f32_16x16x32_bf16(false, (A), false, (B), (short)0, (C), false, false)

// sched_group_barrier masks
#define SG_WMMA      0x008
#define SG_VMEM_READ 0x020
#define SG_DS_READ   0x100

// =====================================================================
// Kernel 1: fp32 -> bf16 weights, pre-swizzled into WMMA B-fragment order.
// B 32x16 (ISA 7.12.2): lane L<16 -> col N=L, K={0..7,16..23};
// lane L>=16 -> col N=L-16, K={8..15,24..31}. 32B/lane contiguous.
// =====================================================================
__global__ void convert_weights(const float* __restrict__ w1,
                                const float* __restrict__ w2,
                                __bf16* __restrict__ w1s,
                                __bf16* __restrict__ w2s) {
    int wave = (blockIdx.x * blockDim.x + threadIdx.x) >> 5;
    int lane = threadIdx.x & 31;
    int half = lane >> 4;
    int cl   = lane & 15;
    v16bf v;
    if (wave < (int)W1S_FRAGS) {
        int n  = wave % 128;
        int kb = (wave / 128) % 32;
        int e  = wave / (128 * 32);
        int c  = n * 16 + cl;
        const float* src = w1 + (size_t)e * HDIM * FDIM + c;   // w1[E,H,F]
#pragma unroll
        for (int j = 0; j < 16; ++j) {
            int kl = ((j < 8) ? j : j + 8) + half * 8;
            int K  = kb * 32 + kl;
            v[j] = f2bf(src[(size_t)K * FDIM]);
        }
        *(v16bf*)(w1s + (size_t)wave * 512 + lane * 16) = v;
    } else {
        int w = wave - (int)W1S_FRAGS;
        if (w >= (int)W2S_FRAGS) return;
        int n  = w % 64;
        int kb = (w / 64) % 64;
        int e  = w / (64 * 64);
        int c  = n * 16 + cl;
        const float* src = w2 + (size_t)e * FDIM * HDIM + c;   // w2[E,F,H]
#pragma unroll
        for (int j = 0; j < 16; ++j) {
            int kl = ((j < 8) ? j : j + 8) + half * 8;
            int K  = kb * 32 + kl;
            v[j] = f2bf(src[(size_t)K * HDIM]);
        }
        *(v16bf*)(w2s + (size_t)w * 512 + lane * 16) = v;
    }
}

// =====================================================================
// Kernel 2: router (wave per token).
// =====================================================================
__global__ void router_kernel(const float* __restrict__ x,
                              const float* __restrict__ wr,
                              int* __restrict__ counts,
                              float* __restrict__ usage,
                              int* __restrict__ tok_list,
                              float* __restrict__ prob_list) {
    int wave = (blockIdx.x * blockDim.x + threadIdx.x) >> 5;
    int lane = threadIdx.x & 31;
    if (wave >= N_TOK) return;
    const float* xr = x + (size_t)wave * HDIM;

    float acc[NEXP];
#pragma unroll
    for (int e = 0; e < NEXP; ++e) acc[e] = 0.f;
    for (int h = lane; h < HDIM; h += 32) {
        float xv = xr[h];
        const float* w = wr + (size_t)h * NEXP;
#pragma unroll
        for (int e = 0; e < NEXP; ++e) acc[e] = fmaf(xv, w[e], acc[e]);
    }
#pragma unroll
    for (int e = 0; e < NEXP; ++e) {
#pragma unroll
        for (int m = 16; m >= 1; m >>= 1) acc[e] += __shfl_xor(acc[e], m, 32);
    }
    float mx = acc[0];
#pragma unroll
    for (int e = 1; e < NEXP; ++e) mx = fmaxf(mx, acc[e]);
    float p[NEXP]; float s = 0.f;
#pragma unroll
    for (int e = 0; e < NEXP; ++e) { p[e] = expf(acc[e] - mx); s += p[e]; }
    float inv = 1.f / s;
#pragma unroll
    for (int e = 0; e < NEXP; ++e) p[e] *= inv;

    int i0 = 0;
#pragma unroll
    for (int e = 1; e < NEXP; ++e) if (p[e] > p[i0]) i0 = e;
    int i1 = (i0 == 0) ? 1 : 0;
#pragma unroll
    for (int e = 0; e < NEXP; ++e) if (e != i0 && p[e] > p[i1]) i1 = e;
    float pn = p[i0] + p[i1];
    float p0 = p[i0] / pn, p1 = p[i1] / pn;

    if (lane < NEXP) atomicAdd(&usage[lane], p[lane]);
    if (lane == 0) {
        int pos0 = atomicAdd(&counts[i0], 1);
        tok_list[i0 * CAP + pos0]  = wave;
        prob_list[i0 * CAP + pos0] = p0;
        int pos1 = atomicAdd(&counts[i1], 1);
        tok_list[i1 * CAP + pos1]  = wave;
        prob_list[i1 * CAP + pos1] = p1;
    }
}

// =====================================================================
// Kernel 3: fused grouped FFN, 32 gathered tokens x 1 expert per block.
// F processed in 8 chunks of 256: h_c = x@w1[:,chunk]+b1 -> GELU -> LDS
// -> y_acc += g_c @ w2[chunk,:]. Persistent y_acc[2][8] tiles per wave.
// Pipeline stages pinned with sched_barrier(0) at every K-iteration
// boundary so the next K-block's B loads cannot be sunk to their
// consumers; group barriers interleave loads with the WMMA chain.
// =====================================================================
__global__ __launch_bounds__(256)
__attribute__((amdgpu_waves_per_eu(2)))
void moe_ffn_kernel(const float* __restrict__ x,
                    const float* __restrict__ b1,
                    const float* __restrict__ b2,
                    const __bf16* __restrict__ w1s,
                    const __bf16* __restrict__ w2s,
                    const int* __restrict__ counts,
                    const int* __restrict__ tok_list,
                    const float* __restrict__ prob_list,
                    float* __restrict__ out) {
    __shared__ __align__(64) __bf16 xfrag[2 * 32 * 32 * 16];   // 64 KB: A-frags of x, K=H
    __shared__ __align__(64) __bf16 gchunk[2 * 8 * 32 * 16];   // 16 KB: A-frags of gelu chunk, K=256
    __shared__ int   s_tok[TILE_M];
    __shared__ float s_prob[TILE_M];

    int e    = blockIdx.y;
    int tile = blockIdx.x;
    int cnt  = counts[e];
    if (tile * TILE_M >= cnt) return;

    int tid = threadIdx.x;
    if (tid < TILE_M) {
        int r = tile * TILE_M + tid;
        if (r < cnt) { s_tok[tid] = tok_list[e * CAP + r]; s_prob[tid] = prob_list[e * CAP + r]; }
        else         { s_tok[tid] = 0;                     s_prob[tid] = 0.f; }
    }
    __syncthreads();

    // ---- gather + convert x tile into A-fragment order ----
    for (int i = tid; i < TILE_M * HDIM; i += 256) {
        int m = i >> 10, h = i & (HDIM - 1);
        float val = x[(size_t)s_tok[m] * HDIM + h];
        int mt = m >> 4, ml = m & 15;
        int kb = h >> 5, kin = h & 31;
        int ln = ml + ((kin >> 3) & 1) * 16;
        int el = (kin & 7) + ((kin >= 16) ? 8 : 0);
        xfrag[mt * 16384 + (kb * 32 + ln) * 16 + el] = f2bf(val);
    }
    __syncthreads();

    int wave = tid >> 5, lane = tid & 31;
    int cl = lane & 15, hf = lane >> 4;

    // persistent FFN2 accumulators: wave owns n-tiles [wave*8, wave*8+8), both M-tiles
    v8f acc2[2][8];
#pragma unroll
    for (int j = 0; j < 8; ++j) {
        float bv = b2[e * HDIM + (wave * 8 + j) * 16 + cl];
#pragma unroll
        for (int r = 0; r < 8; ++r) { acc2[0][j][r] = bv; acc2[1][j][r] = bv; }
    }

    for (int c = 0; c < 8; ++c) {
        // warm L2/L0 for next chunk's w1 slice (global_prefetch_b8)
        if (c + 1 < 8)
            __builtin_prefetch(w1s + ((size_t)(e * 32) * 128 + ((c + 1) * 16 + wave * 2)) * 512 + lane * 16, 0, 1);

        // ---------- stage 1: h_c[32, 256] = x @ w1[:, c*256 : +256] ----------
        // wave owns chunk n-tiles {wave*2, wave*2+1}, both M-tiles
        v8f acc1[2][2];
#pragma unroll
        for (int j = 0; j < 2; ++j) {
            float bv = b1[e * FDIM + c * 256 + (wave * 2 + j) * 16 + cl];
#pragma unroll
            for (int r = 0; r < 8; ++r) { acc1[0][j][r] = bv; acc1[1][j][r] = bv; }
        }
        {
            // fragment (kb, j) at wb + kb*65536 + j*512 elements; kb-pair double buffer
            const __bf16* wb = w1s + ((size_t)(e * 32) * 128 + (c * 16 + wave * 2)) * 512 + lane * 16;
            v16bf B[2][4];
#pragma unroll
            for (int q = 0; q < 2; ++q) {
                B[0][q * 2 + 0] = *(const v16bf*)(wb + (size_t)q * 65536);
                B[0][q * 2 + 1] = *(const v16bf*)(wb + (size_t)q * 65536 + 512);
            }
            __builtin_amdgcn_sched_barrier(0);
#pragma unroll
            for (int kp = 0; kp < 16; ++kp) {
                const int cur = kp & 1, nxt = cur ^ 1;
                if (kp + 1 < 16) {
                    const __bf16* wn = wb + (size_t)(kp + 1) * 131072;
#pragma unroll
                    for (int q = 0; q < 2; ++q) {
                        B[nxt][q * 2 + 0] = *(const v16bf*)(wn + (size_t)q * 65536);
                        B[nxt][q * 2 + 1] = *(const v16bf*)(wn + (size_t)q * 65536 + 512);
                    }
                }
#pragma unroll
                for (int q = 0; q < 2; ++q) {
                    int kb = kp * 2 + q;
                    v16bf a0 = *(const v16bf*)&xfrag[(kb * 32 + lane) * 16];
                    v16bf a1 = *(const v16bf*)&xfrag[16384 + (kb * 32 + lane) * 16];
                    acc1[0][0] = WMMA_BF16(a0, B[cur][q * 2 + 0], acc1[0][0]);
                    acc1[1][0] = WMMA_BF16(a1, B[cur][q * 2 + 0], acc1[1][0]);
                    acc1[0][1] = WMMA_BF16(a0, B[cur][q * 2 + 1], acc1[0][1]);
                    acc1[1][1] = WMMA_BF16(a1, B[cur][q * 2 + 1], acc1[1][1]);
                }
                // interleave within the iteration, then pin the boundary
                __builtin_amdgcn_sched_group_barrier(SG_DS_READ, 8, 0);
#pragma unroll
                for (int g = 0; g < 4; ++g) {
                    __builtin_amdgcn_sched_group_barrier(SG_VMEM_READ, 2, 0);
                    __builtin_amdgcn_sched_group_barrier(SG_WMMA, 2, 0);
                }
                __builtin_amdgcn_sched_barrier(0);   // no cross-iteration motion
            }
        }
        // GELU (exact) -> gchunk A-fragments (K-local = 256)
#pragma unroll
        for (int mt = 0; mt < 2; ++mt) {
#pragma unroll
            for (int j = 0; j < 2; ++j) {
                int ccol = (wave * 2 + j) * 16 + cl;         // column within chunk [0,256)
                int kb2 = ccol >> 5, kin = ccol & 31;
                int lbit = ((kin >> 3) & 1) * 16;
                int el   = (kin & 7) + ((kin >= 16) ? 8 : 0);
#pragma unroll
                for (int r = 0; r < 8; ++r) {
                    int M = r + hf * 8;
                    float hv = acc1[mt][j][r];
                    float g  = 0.5f * hv * (1.0f + erff(hv * 0.70710678118f));
                    gchunk[mt * 4096 + (kb2 * 32 + (M + lbit)) * 16 + el] = f2bf(g);
                }
            }
        }
        __syncthreads();

        // ---------- stage 2: y_acc += g_c[32,256] @ w2[c*256 : +256, :] ----------
        {
            // fragment (kb, j) at wb2 + kb*32768 + j*512 elements; kb double buffer
            const __bf16* wb2 = w2s + ((size_t)(e * 64 + c * 8) * 64 + wave * 8) * 512 + lane * 16;
            v16bf B[2][8];
#pragma unroll
            for (int j = 0; j < 8; ++j)
                B[0][j] = *(const v16bf*)(wb2 + (size_t)j * 512);
            __builtin_amdgcn_sched_barrier(0);
#pragma unroll
            for (int kb = 0; kb < 8; ++kb) {
                const int cur = kb & 1, nxt = cur ^ 1;
                if (kb + 1 < 8) {
                    const __bf16* wn = wb2 + (size_t)(kb + 1) * 32768;
#pragma unroll
                    for (int j = 0; j < 8; ++j)
                        B[nxt][j] = *(const v16bf*)(wn + (size_t)j * 512);
                }
                v16bf a0 = *(const v16bf*)&gchunk[(kb * 32 + lane) * 16];
                v16bf a1 = *(const v16bf*)&gchunk[4096 + (kb * 32 + lane) * 16];
#pragma unroll
                for (int j = 0; j < 8; ++j) {
                    acc2[0][j] = WMMA_BF16(a0, B[cur][j], acc2[0][j]);
                    acc2[1][j] = WMMA_BF16(a1, B[cur][j], acc2[1][j]);
                }
                // interleave within the iteration, then pin the boundary
                __builtin_amdgcn_sched_group_barrier(SG_DS_READ, 4, 0);
#pragma unroll
                for (int g = 0; g < 8; ++g) {
                    __builtin_amdgcn_sched_group_barrier(SG_VMEM_READ, 2, 0);
                    __builtin_amdgcn_sched_group_barrier(SG_WMMA, 2, 0);
                }
                __builtin_amdgcn_sched_barrier(0);   // no cross-iteration motion
            }
        }
        __syncthreads();   // gchunk reused next chunk
    }

    // ---------- epilogue: weighted atomic scatter ----------
#pragma unroll
    for (int mt = 0; mt < 2; ++mt) {
#pragma unroll
        for (int j = 0; j < 8; ++j) {
            int col = (wave * 8 + j) * 16 + cl;
#pragma unroll
            for (int r = 0; r < 8; ++r) {
                int m = mt * 16 + r + hf * 8;
                atomicAdd(&out[(size_t)s_tok[m] * HDIM + col], s_prob[m] * acc2[mt][j][r]);
            }
        }
    }
}

// =====================================================================
// Kernel 4: aux load-balance loss.
// =====================================================================
__global__ void aux_kernel(const float* __restrict__ usage, float* __restrict__ out_aux) {
    if (threadIdx.x == 0) {
        float s = 0.f;
#pragma unroll
        for (int e = 0; e < NEXP; ++e) {
            float u = usage[e] * (1.0f / (float)N_TOK) - 0.125f;
            s += u * u;
        }
        *out_aux = 0.01f * s;
    }
}

// =====================================================================
extern "C" void kernel_launch(void* const* d_in, const int* in_sizes, int n_in,
                              void* d_out, int out_size, void* d_ws, size_t ws_size,
                              hipStream_t stream) {
    const float* x        = (const float*)d_in[0];
    const float* w_router = (const float*)d_in[1];
    const float* w1       = (const float*)d_in[2];
    const float* b1       = (const float*)d_in[3];
    const float* w2       = (const float*)d_in[4];
    const float* b2       = (const float*)d_in[5];
    float* out = (float*)d_out;

    char* ws = (char*)d_ws;
    int*    counts    = (int*)(ws + WS_COUNTS);
    float*  usage     = (float*)(ws + WS_USAGE);
    int*    tok_list  = (int*)(ws + WS_TOK);
    float*  prob_list = (float*)(ws + WS_PROB);
    __bf16* w1s       = (__bf16*)(ws + WS_W1S);
    __bf16* w2s       = (__bf16*)(ws + WS_W2S);

    hipMemsetAsync(d_ws, 0, 1024, stream);
    hipMemsetAsync(d_out, 0, (size_t)N_TOK * HDIM * sizeof(float), stream);

    convert_weights<<<8192, 256, 0, stream>>>(w1, w2, w1s, w2s);
    router_kernel<<<N_TOK / 8, 256, 0, stream>>>(x, w_router, counts, usage, tok_list, prob_list);
    moe_ffn_kernel<<<dim3(CAP / TILE_M, NEXP), 256, 0, stream>>>(
        x, b1, b2, w1s, w2s, counts, tok_list, prob_list, out);
    aux_kernel<<<1, 32, 0, stream>>>(usage, out + (size_t)N_TOK * HDIM);
}
